// HeteroConvFraudDetector_90804198572575
// MI455X (gfx1250) — matrix operand order repaired and checked by reading, not compiled, not run on hardware
//
#include <hip/hip_runtime.h>

#define N_ENT 50000
#define N_TX  40000
#define D     128
#define FDIM  432
#define T     11
#define KCAT  1840      // FDIM + T*D
#define KT1   58        // ceil(1840/32)
#define KCHUNK 16       // K-tiles staged per LDS phase in cls1

typedef __attribute__((ext_vector_type(16))) __bf16 v16bf;
typedef __attribute__((ext_vector_type(8)))  float  v8f;

// ---------------- helpers ----------------

__device__ __forceinline__ unsigned short f2bf_bits(float f) {
    unsigned u = __float_as_uint(f);
    unsigned r = u + 0x7FFFu + ((u >> 16) & 1u);   // round-to-nearest-even
    return (unsigned short)(r >> 16);
}

__device__ __forceinline__ void atomicMaxFloat(float* addr, float val) {
    int old = __float_as_int(*addr);
    while (val > __int_as_float(old)) {
        int assumed = old;
        old = atomicCAS((int*)addr, assumed, __float_as_int(val));
        if (old == assumed) break;
    }
}

// ---------------- generic utility kernels ----------------

__global__ void zero_kernel(float* p, size_t n) {
    size_t i = (size_t)blockIdx.x * blockDim.x + threadIdx.x;
    size_t stride = (size_t)gridDim.x * blockDim.x;
    for (; i < n; i += stride) p[i] = 0.0f;
}

__global__ void init_seg_kernel(float* m, float* den, int n) {
    int i = blockIdx.x * blockDim.x + threadIdx.x;
    if (i < n) { m[i] = -__builtin_inff(); den[i] = 0.0f; }
}

// Pack a row-major f32 weight [K x N] into the WMMA bf16 B-fragment layout:
// fragment (kt, nt): lane L holds column N = nt*16 + (L&15),
// halves p=0..15 hold K = kt*32 + (L>=16 ? 8 : 0) + (p&7) + (p>=8 ? 16 : 0).  (zero-padded past K)
__global__ void pack_b_kernel(const float* __restrict__ W, unsigned short* __restrict__ P,
                              int K, int N, int KT, int NT) {
    int i = blockIdx.x * blockDim.x + threadIdx.x;
    int total = KT * NT * 512;
    if (i >= total) return;
    int p    = i & 15;
    int lane = (i >> 4) & 31;
    int frag = i >> 9;
    int nt   = frag % NT;
    int kt   = frag / NT;
    int k = kt * 32 + ((lane >> 4) ? 8 : 0) + (p & 7) + ((p >> 3) << 4);
    int n = nt * 16 + (lane & 15);
    float v = (k < K && n < N) ? W[(size_t)k * N + n] : 0.0f;
    P[i] = f2bf_bits(v);
}

// ---------------- projection: q,k,v,skip = gather(h) @ W + b ----------------
// grid = N_ENT/16 blocks, 128 threads (4 waves; wave w computes matrix w).
// A tile staged once per block into LDS in WMMA A-fragment layout (bf16).

__global__ __launch_bounds__(128)
void proj_kernel(const float* __restrict__ emb_t, const int* __restrict__ idx,
                 const unsigned short* __restrict__ pWq, const unsigned short* __restrict__ pWk,
                 const unsigned short* __restrict__ pWv, const unsigned short* __restrict__ pWs,
                 const float* __restrict__ bq, const float* __restrict__ bk,
                 const float* __restrict__ bv, const float* __restrict__ bs,
                 float* __restrict__ qo, float* __restrict__ ko,
                 float* __restrict__ vo, float* __restrict__ so) {
    __shared__ __align__(32) unsigned short aLds[4 * 32 * 16];  // 4 kt-tiles, 4 KB

    int tid = threadIdx.x;
    int wave = tid >> 5;
    int lane = tid & 31;
    int laneHi = lane >> 4;
    int mtile = blockIdx.x;

    // ---- cooperative A staging: thread t fills A-fragment slot (kt=t>>5, slane=t&31)
    {
        int skt = tid >> 5;
        int sl  = tid & 31;
        int srow = mtile * 16 + (sl & 15);
        const float* hrow = emb_t + (size_t)idx[srow] * D;
        int c0 = skt * 32 + (sl >> 4) * 8;   // A layout: K 0..7/8..15, then 16..23/24..31
        unsigned short* dp = &aLds[((skt << 5) + sl) * 16];
        #pragma unroll
        for (int i = 0; i < 8; ++i) dp[i]     = f2bf_bits(hrow[c0 + i]);
        #pragma unroll
        for (int i = 0; i < 8; ++i) dp[8 + i] = f2bf_bits(hrow[c0 + 16 + i]);
    }
    __syncthreads();

    const unsigned short* pW = (wave == 0) ? pWq : (wave == 1) ? pWk : (wave == 2) ? pWv : pWs;
    const float* bias = (wave == 0) ? bq : (wave == 1) ? bk : (wave == 2) ? bv : bs;
    float* out = (wave == 0) ? qo : (wave == 1) ? ko : (wave == 2) ? vo : so;

    v8f acc[8] = {};
    #pragma unroll
    for (int kt = 0; kt < 4; ++kt) {
        v16bf a = *(const v16bf*)&aLds[((kt << 5) + lane) * 16];
        v16bf bfrag[8];
        #pragma unroll
        for (int nt = 0; nt < 8; ++nt)
            bfrag[nt] = *(const v16bf*)(pW + ((size_t)(kt * 8 + nt) * 32 + lane) * 16);
        #pragma unroll
        for (int nt = 0; nt < 8; ++nt)
            acc[nt] = __builtin_amdgcn_wmma_f32_16x16x32_bf16(
                false, a, false, bfrag[nt], (short)0, acc[nt], false, false);
    }
    #pragma unroll
    for (int nt = 0; nt < 8; ++nt) {
        int col = nt * 16 + (lane & 15);
        float bv_ = bias[col];
        #pragma unroll
        for (int j = 0; j < 8; ++j) {
            int r = mtile * 16 + j + laneHi * 8;   // C layout per ISA
            out[(size_t)r * D + col] = acc[nt][j] + bv_;
        }
    }
}

// ---------------- attention ----------------

__global__ __launch_bounds__(256)
void score_kernel(const float* __restrict__ q, const float* __restrict__ k,
                  const int* __restrict__ src, const int* __restrict__ dst,
                  float* __restrict__ scores, float* __restrict__ m, int nedge) {
    int e = blockIdx.x * 8 + (threadIdx.x >> 5);
    int lane = threadIdx.x & 31;
    if (e >= nedge) return;
    const float* qr = q + (size_t)dst[e] * D;
    const float* kr = k + (size_t)src[e] * D;
    float s = 0.0f;
    #pragma unroll
    for (int i = 0; i < 4; ++i) { int c = lane + 32 * i; s += qr[c] * kr[c]; }
    for (int o = 16; o; o >>= 1) s += __shfl_xor(s, o, 32);
    if (lane == 0) {
        float sc = s * 0.08838834764831845f;  // 1/sqrt(128)
        scores[e] = sc;
        atomicMaxFloat(&m[dst[e]], sc);
    }
}

__global__ void exp_kernel(const float* __restrict__ scores, const float* __restrict__ m,
                           const int* __restrict__ dst, float* __restrict__ ex,
                           float* __restrict__ den, int nedge) {
    int e = blockIdx.x * blockDim.x + threadIdx.x;
    if (e >= nedge) return;
    float v = __expf(scores[e] - m[dst[e]]);
    ex[e] = v;
    atomicAdd(&den[dst[e]], v);
}

__global__ __launch_bounds__(256)
void agg_kernel(const float* __restrict__ ex, const float* __restrict__ den,
                const float* __restrict__ vmat, const int* __restrict__ src,
                const int* __restrict__ dst, float* __restrict__ outb, int nedge) {
    int e = blockIdx.x * 8 + (threadIdx.x >> 5);
    int lane = threadIdx.x & 31;
    if (e >= nedge) return;
    float alpha = ex[e] / den[dst[e]];
    const float* vr = vmat + (size_t)src[e] * D;
    float* orow = outb + (size_t)dst[e] * D;
    #pragma unroll
    for (int i = 0; i < 4; ++i) { int c = lane + 32 * i; atomicAdd(&orow[c], alpha * vr[c]); }
}

__global__ __launch_bounds__(256)
void ln_kernel(float* __restrict__ outb, const float* __restrict__ g,
               const float* __restrict__ b, int nrows) {
    int r = blockIdx.x * 8 + (threadIdx.x >> 5);
    int lane = threadIdx.x & 31;
    if (r >= nrows) return;
    float* row = outb + (size_t)r * D;
    float x0 = row[lane], x1 = row[lane + 32], x2 = row[lane + 64], x3 = row[lane + 96];
    float s = x0 + x1 + x2 + x3;
    for (int o = 16; o; o >>= 1) s += __shfl_xor(s, o, 32);
    float mu = s * (1.0f / 128.0f);
    float d0 = x0 - mu, d1 = x1 - mu, d2 = x2 - mu, d3 = x3 - mu;
    float ss = d0 * d0 + d1 * d1 + d2 * d2 + d3 * d3;
    for (int o = 16; o; o >>= 1) ss += __shfl_xor(ss, o, 32);
    float inv = rsqrtf(ss * (1.0f / 128.0f) + 1e-5f);
    row[lane]      = d0 * inv * g[lane]      + b[lane];
    row[lane + 32] = d1 * inv * g[lane + 32] + b[lane + 32];
    row[lane + 64] = d2 * inv * g[lane + 64] + b[lane + 64];
    row[lane + 96] = d3 * inv * g[lane + 96] + b[lane + 96];
}

__global__ __launch_bounds__(256)
void scatter_kernel(const float* __restrict__ hn, const int* __restrict__ src,
                    const int* __restrict__ dst, float* __restrict__ agg, int t, int nedge) {
    int e = blockIdx.x * 8 + (threadIdx.x >> 5);
    int lane = threadIdx.x & 31;
    if (e >= nedge) return;
    const float* hr = hn + (size_t)src[e] * D;
    float* ar = agg + (size_t)dst[e] * (T * D) + t * D;
    #pragma unroll
    for (int i = 0; i < 4; ++i) { int c = lane + 32 * i; atomicAdd(&ar[c], hr[c]); }
}

// ---------------- classifier ----------------
// x1 = relu([tx_x | agg] @ W1 + b1) : K = 1840 (padded to 1856), N = 128.
// grid = N_TX/16, 128 threads, wave w covers ntiles {2w, 2w+1}.
// A staged in LDS in KCHUNK-sized K-tile phases (bf16 fragment layout).

__global__ __launch_bounds__(128)
void cls1_kernel(const float* __restrict__ tx, const float* __restrict__ agg,
                 const unsigned short* __restrict__ pW1, const float* __restrict__ b1,
                 float* __restrict__ x1) {
    __shared__ __align__(32) unsigned short aLds[KCHUNK * 32 * 16];  // 16 KB

    int tid = threadIdx.x;
    int wave = tid >> 5;
    int lane = tid & 31;
    int laneHi = lane >> 4;
    int mtile = blockIdx.x;

    v8f acc[2] = {};
    for (int kt0 = 0; kt0 < KT1; kt0 += KCHUNK) {
        int ktn = (kt0 + KCHUNK < KT1) ? KCHUNK : (KT1 - kt0);
        int nslots = ktn * 32;
        // ---- cooperative A staging for this K-chunk
        for (int s = tid; s < nslots; s += 128) {
            int skt = kt0 + (s >> 5);
            int sl  = s & 31;
            int srow = mtile * 16 + (sl & 15);
            const float* txr = tx + (size_t)srow * FDIM;
            const float* agr = agg + (size_t)srow * (T * D);
            int c0 = skt * 32 + (sl >> 4) * 8;
            unsigned short* dp = &aLds[s * 16];
            #pragma unroll
            for (int i = 0; i < 8; ++i) {
                int kk = c0 + i;
                float f = (kk < FDIM) ? txr[kk] : ((kk < KCAT) ? agr[kk - FDIM] : 0.0f);
                dp[i] = f2bf_bits(f);
            }
            #pragma unroll
            for (int i = 0; i < 8; ++i) {
                int kk = c0 + 16 + i;
                float f = (kk < FDIM) ? txr[kk] : ((kk < KCAT) ? agr[kk - FDIM] : 0.0f);
                dp[8 + i] = f2bf_bits(f);
            }
        }
        __syncthreads();
        // ---- WMMA over this K-chunk
        #pragma unroll 2
        for (int kt = kt0; kt < kt0 + ktn; ++kt) {
            v16bf a = *(const v16bf*)&aLds[(((kt - kt0) << 5) + lane) * 16];
            v16bf b0 = *(const v16bf*)(pW1 + ((size_t)(kt * 8 + wave * 2 + 0) * 32 + lane) * 16);
            v16bf b1f = *(const v16bf*)(pW1 + ((size_t)(kt * 8 + wave * 2 + 1) * 32 + lane) * 16);
            if (kt + 1 < KT1) {  // prefetch next K-tile's fragments for this wave
                __builtin_prefetch((const void*)(pW1 + ((size_t)((kt + 1) * 8 + wave * 2) * 32 + lane) * 16), 0, 0);
            }
            acc[0] = __builtin_amdgcn_wmma_f32_16x16x32_bf16(
                false, a, false, b0, (short)0, acc[0], false, false);
            acc[1] = __builtin_amdgcn_wmma_f32_16x16x32_bf16(
                false, a, false, b1f, (short)0, acc[1], false, false);
        }
        __syncthreads();
    }
    #pragma unroll
    for (int u = 0; u < 2; ++u) {
        int nt = wave * 2 + u;
        int col = nt * 16 + (lane & 15);
        float bias = b1[col];
        #pragma unroll
        for (int j = 0; j < 8; ++j) {
            int r = mtile * 16 + j + laneHi * 8;
            float v = acc[u][j] + bias;
            x1[(size_t)r * D + col] = v > 0.0f ? v : 0.0f;
        }
    }
}

// x2 = relu(x1 @ W2 + b2) : [40000,128] @ [128,64]. wave = ntile (4 ntiles).
__global__ __launch_bounds__(128)
void cls2_kernel(const float* __restrict__ x1, const unsigned short* __restrict__ pW2,
                 const float* __restrict__ b2, float* __restrict__ x2) {
    __shared__ __align__(32) unsigned short aLds[4 * 32 * 16];  // 4 KB

    int tid = threadIdx.x;
    int nt = tid >> 5;
    int lane = tid & 31;
    int laneHi = lane >> 4;
    int mtile = blockIdx.x;

    {
        int skt = tid >> 5;
        int sl  = tid & 31;
        int srow = mtile * 16 + (sl & 15);
        const float* xr = x1 + (size_t)srow * D;
        int c0 = skt * 32 + (sl >> 4) * 8;
        unsigned short* dp = &aLds[((skt << 5) + sl) * 16];
        #pragma unroll
        for (int i = 0; i < 8; ++i) dp[i]     = f2bf_bits(xr[c0 + i]);
        #pragma unroll
        for (int i = 0; i < 8; ++i) dp[8 + i] = f2bf_bits(xr[c0 + 16 + i]);
    }
    __syncthreads();

    v8f acc[1] = {};
    v16bf bfrag[4];
    #pragma unroll
    for (int kt = 0; kt < 4; ++kt)
        bfrag[kt] = *(const v16bf*)(pW2 + ((size_t)(kt * 4 + nt) * 32 + lane) * 16);
    #pragma unroll
    for (int kt = 0; kt < 4; ++kt) {
        v16bf a = *(const v16bf*)&aLds[((kt << 5) + lane) * 16];
        acc[0] = __builtin_amdgcn_wmma_f32_16x16x32_bf16(
            false, a, false, bfrag[kt], (short)0, acc[0], false, false);
    }
    int col = nt * 16 + (lane & 15);
    float bias = b2[col];
    #pragma unroll
    for (int j = 0; j < 8; ++j) {
        int r = mtile * 16 + j + laneHi * 8;
        float v = acc[0][j] + bias;
        x2[(size_t)r * 64 + col] = v > 0.0f ? v : 0.0f;
    }
}

__global__ void cls3_kernel(const float* __restrict__ x2, const float* __restrict__ W3,
                            const float* __restrict__ b3, float* __restrict__ out, int n) {
    int r = blockIdx.x * blockDim.x + threadIdx.x;
    if (r >= n) return;
    float s = b3[0];
    const float* xr = x2 + (size_t)r * 64;
    #pragma unroll 8
    for (int i = 0; i < 64; ++i) s += xr[i] * W3[i];
    out[r] = s;
}

// ---------------- host launch ----------------

extern "C" void kernel_launch(void* const* d_in, const int* in_sizes, int n_in,
                              void* d_out, int out_size, void* d_ws, size_t ws_size,
                              hipStream_t stream) {
    (void)in_sizes; (void)n_in; (void)out_size; (void)ws_size;

    const float* tx_x     = (const float*)d_in[0];
    const int*   entity_x = (const int*)d_in[1];
    const int*   edge_src = (const int*)d_in[2];
    const int*   edge_dst = (const int*)d_in[3];
    const float* emb  = (const float*)d_in[4];
    const float* ln_g = (const float*)d_in[5];
    const float* ln_b = (const float*)d_in[6];
    const float* Wq = (const float*)d_in[7];   const float* bq = (const float*)d_in[8];
    const float* Wk = (const float*)d_in[9];   const float* bk = (const float*)d_in[10];
    const float* Wv = (const float*)d_in[11];  const float* bv = (const float*)d_in[12];
    const float* Ws = (const float*)d_in[13];  const float* bs = (const float*)d_in[14];
    const float* W1 = (const float*)d_in[15];  const float* b1 = (const float*)d_in[16];
    const float* W2 = (const float*)d_in[17];  const float* b2 = (const float*)d_in[18];
    const float* W3 = (const float*)d_in[19];  const float* b3 = (const float*)d_in[20];
    float* logits = (float*)d_out;

    char* ws = (char*)d_ws;
    size_t off = 0;
    auto carve = [&](size_t bytes) -> char* {
        char* p = ws + off;
        off = (off + bytes + 255) & ~(size_t)255;
        return p;
    };

    unsigned short* pWq = (unsigned short*)carve((size_t)4 * 8 * 512 * 2);
    unsigned short* pWk = (unsigned short*)carve((size_t)4 * 8 * 512 * 2);
    unsigned short* pWv = (unsigned short*)carve((size_t)4 * 8 * 512 * 2);
    unsigned short* pWs = (unsigned short*)carve((size_t)4 * 8 * 512 * 2);
    unsigned short* pW1 = (unsigned short*)carve((size_t)KT1 * 8 * 512 * 2);
    unsigned short* pW2 = (unsigned short*)carve((size_t)4 * 4 * 512 * 2);
    float* qbuf  = (float*)carve((size_t)N_ENT * D * 4);
    float* kbuf  = (float*)carve((size_t)N_ENT * D * 4);
    float* vbuf  = (float*)carve((size_t)N_ENT * D * 4);
    float* obuf  = (float*)carve((size_t)N_ENT * D * 4);
    float* sc    = (float*)carve((size_t)N_TX * 4);
    float* exb   = (float*)carve((size_t)N_TX * 4);
    float* mbuf  = (float*)carve((size_t)N_ENT * 4);
    float* dbuf  = (float*)carve((size_t)N_ENT * 4);
    float* aggb  = (float*)carve((size_t)N_TX * T * D * 4);
    float* x1b   = (float*)carve((size_t)N_TX * D * 4);
    float* x2b   = (float*)carve((size_t)N_TX * 64 * 4);

    // zero the aggregation tensor, pack weights to bf16 WMMA fragments
    zero_kernel<<<2048, 256, 0, stream>>>(aggb, (size_t)N_TX * T * D);
    int tp = 4 * 8 * 512;
    pack_b_kernel<<<(tp + 255) / 256, 256, 0, stream>>>(Wq, pWq, 128, 128, 4, 8);
    pack_b_kernel<<<(tp + 255) / 256, 256, 0, stream>>>(Wk, pWk, 128, 128, 4, 8);
    pack_b_kernel<<<(tp + 255) / 256, 256, 0, stream>>>(Wv, pWv, 128, 128, 4, 8);
    pack_b_kernel<<<(tp + 255) / 256, 256, 0, stream>>>(Ws, pWs, 128, 128, 4, 8);
    int t1 = KT1 * 8 * 512;
    pack_b_kernel<<<(t1 + 255) / 256, 256, 0, stream>>>(W1, pW1, KCAT, 128, KT1, 8);
    int t2 = 4 * 4 * 512;
    pack_b_kernel<<<(t2 + 255) / 256, 256, 0, stream>>>(W2, pW2, 128, 64, 4, 4);

    for (int t = 0; t < T; ++t) {
        const float* emb_t = emb + (size_t)t * N_ENT * D;
        const int* idx = entity_x + (size_t)t * N_ENT;
        const int* src = edge_src + (size_t)t * N_TX;
        const int* dst = edge_dst + (size_t)t * N_TX;

        proj_kernel<<<N_ENT / 16, 128, 0, stream>>>(
            emb_t, idx, pWq, pWk, pWv, pWs, bq, bk, bv, bs, qbuf, kbuf, vbuf, obuf);
        init_seg_kernel<<<(N_ENT + 255) / 256, 256, 0, stream>>>(mbuf, dbuf, N_ENT);
        score_kernel<<<N_TX / 8, 256, 0, stream>>>(qbuf, kbuf, src, dst, sc, mbuf, N_TX);
        exp_kernel<<<(N_TX + 255) / 256, 256, 0, stream>>>(sc, mbuf, dst, exb, dbuf, N_TX);
        agg_kernel<<<N_TX / 8, 256, 0, stream>>>(exb, dbuf, vbuf, src, dst, obuf, N_TX);
        ln_kernel<<<N_ENT / 8, 256, 0, stream>>>(obuf, ln_g + (size_t)t * D,
                                                 ln_b + (size_t)t * D, N_ENT);
        scatter_kernel<<<N_TX / 8, 256, 0, stream>>>(obuf, src, dst, aggb, t, N_TX);
    }

    cls1_kernel<<<N_TX / 16, 128, 0, stream>>>(tx_x, aggb, pW1, b1, x1b);
    cls2_kernel<<<N_TX / 16, 128, 0, stream>>>(x1b, pW2, b2, x2b);
    cls3_kernel<<<(N_TX + 255) / 256, 256, 0, stream>>>(x2b, W3, b3, logits, N_TX);
}
